// SpatialAttention_2388001817023
// MI455X (gfx1250) — matrix operand order repaired and checked by reading, not compiled, not run on hardware
//
#include <hip/hip_runtime.h>
#include <hip/hip_bf16.h>

typedef __attribute__((ext_vector_type(16))) _Float16 v16h;
typedef __attribute__((ext_vector_type(8)))  _Float16 v8h;
typedef __attribute__((ext_vector_type(8)))  float    v8f;

#define NPIX   4096   // H*W = N
#define DHEAD  32
#define HEADS  2
#define LOG2E  1.4426950408889634f
#define LRELU(v) ((v) >= 0.f ? (v) : 0.05f * (v))

// ---------------------------------------------------------------------------
// Stage 1: depthwise Conv2d(C,3C,3,groups=C) + LeakyReLU + InstanceNorm.
// One block per output channel ch = c*3 + e (e: 0=q,1=k,2=v).
// Output written as head-major planes [h][dd][N], h=c>>5, dd=c&31.
// ---------------------------------------------------------------------------
__global__ __launch_bounds__(256) void qkv_kernel(
    const float* __restrict__ src,   // [64][64][64]
    const float* __restrict__ w,     // [192][1][3][3]
    const float* __restrict__ bias,  // [192]
    const float* __restrict__ gw,    // in_w [192]
    const float* __restrict__ gb,    // in_b [192]
    float* __restrict__ qdst, float* __restrict__ kdst, float* __restrict__ vdst)
{
  const int ch = blockIdx.x;            // 0..191
  const int c = ch / 3, e = ch % 3;
  float* dst = (e == 0) ? qdst : (e == 1 ? kdst : vdst);
  if (dst == nullptr) return;           // ir's q is never used

  const float* plane = src + c * NPIX;
  const int tid = threadIdx.x;

  float wk[9];
#pragma unroll
  for (int i = 0; i < 9; i++) wk[i] = w[ch * 9 + i];
  const float bv = bias[ch];

  float val[16];
  float sum = 0.f, ssq = 0.f;
#pragma unroll 4
  for (int j = 0; j < 16; j++) {
    int pix = tid + j * 256;
    int y = pix >> 6, x = pix & 63;
    float acc = bv;
#pragma unroll
    for (int ky = 0; ky < 3; ky++) {
      int iy = y + ky - 1;
      if (iy < 0 || iy > 63) continue;
#pragma unroll
      for (int kx = 0; kx < 3; kx++) {
        int ix = x + kx - 1;
        if (ix < 0 || ix > 63) continue;
        acc += wk[ky * 3 + kx] * plane[iy * 64 + ix];
      }
    }
    acc = LRELU(acc);
    val[j] = acc;
    sum += acc;
    ssq += acc * acc;
  }

  __shared__ float red0[256], red1[256];
  red0[tid] = sum; red1[tid] = ssq;
  __syncthreads();
  for (int s = 128; s > 0; s >>= 1) {
    if (tid < s) { red0[tid] += red0[tid + s]; red1[tid] += red1[tid + s]; }
    __syncthreads();
  }
  const float mean = red0[0] * (1.f / (float)NPIX);
  const float var  = red1[0] * (1.f / (float)NPIX) - mean * mean;
  const float istd = rsqrtf(var + 1e-5f);
  const float sc = istd * gw[ch];
  const float sh = gb[ch] - mean * sc;

  const int h = c >> 5, dd = c & 31;
  float* o = dst + h * DHEAD * NPIX + dd * NPIX;
#pragma unroll 4
  for (int j = 0; j < 16; j++) o[tid + j * 256] = val[j] * sc + sh;
}

// ---------------------------------------------------------------------------
// Stage 2: Conv2d(4,2,3,pad=1) over the (32,4096) image + LeakyReLU.
// ---------------------------------------------------------------------------
__global__ __launch_bounds__(256) void fuse_kernel(
    const float* __restrict__ a,
    const float* __restrict__ bsrc,
    const float* __restrict__ w,     // [2][4][3][3]
    const float* __restrict__ bias,  // [2]
    float* __restrict__ dst)         // [2][32][4096]
{
  int idx = blockIdx.x * 256 + threadIdx.x;        // 0 .. 2*32*4096-1
  int o = idx / (DHEAD * NPIX);
  int rem = idx - o * DHEAD * NPIX;
  int y = rem >> 12, x = rem & 4095;
  float acc = bias[o];
#pragma unroll
  for (int ic = 0; ic < 4; ic++) {
    const float* p = (ic < 2) ? (a + ic * DHEAD * NPIX)
                              : (bsrc + (ic - 2) * DHEAD * NPIX);
#pragma unroll
    for (int ky = 0; ky < 3; ky++) {
      int iy = y + ky - 1;
      if (iy < 0 || iy > 31) continue;
#pragma unroll
      for (int kx = 0; kx < 3; kx++) {
        int ix = x + kx - 1;
        if (ix < 0 || ix > 4095) continue;
        acc += w[((o * 4 + ic) * 3 + ky) * 3 + kx] * p[iy * NPIX + ix];
      }
    }
  }
  dst[idx] = LRELU(acc);
}

// ---------------------------------------------------------------------------
// Stage 3 prep: L2-normalize Q,K over d; convert to f16.
//   Q16,K16: [h][n][32] row-major; V16: [h][32][n] plane-major.
// ---------------------------------------------------------------------------
__global__ __launch_bounds__(256) void prep_kernel(
    const float* __restrict__ q, const float* __restrict__ k,
    const float* __restrict__ v,
    _Float16* __restrict__ Q16, _Float16* __restrict__ K16,
    _Float16* __restrict__ V16)
{
  int idx = blockIdx.x * 256 + threadIdx.x;   // 0..8191
  int h = idx >> 12, n = idx & 4095;
  const float* qp = q + h * DHEAD * NPIX + n;
  const float* kp = k + h * DHEAD * NPIX + n;
  const float* vp = v + h * DHEAD * NPIX + n;

  float qr[DHEAD], kr[DHEAD];
  float qs = 0.f, ks = 0.f;
#pragma unroll
  for (int d = 0; d < DHEAD; d++) {
    float a = qp[d * NPIX]; qr[d] = a; qs += a * a;
    float c = kp[d * NPIX]; kr[d] = c; ks += c * c;
  }
  float qi = 1.f / fmaxf(sqrtf(qs), 1e-12f);
  float ki = 1.f / fmaxf(sqrtf(ks), 1e-12f);

  _Float16 qh16[DHEAD], kh16[DHEAD];
#pragma unroll
  for (int d = 0; d < DHEAD; d++) {
    qh16[d] = (_Float16)(qr[d] * qi);
    kh16[d] = (_Float16)(kr[d] * ki);
  }
  _Float16* qo = Q16 + (h * NPIX + n) * DHEAD;   // 64B aligned
  _Float16* ko = K16 + (h * NPIX + n) * DHEAD;
#pragma unroll
  for (int j = 0; j < 4; j++) {
    *(v8h*)(qo + j * 8) = *(const v8h*)(qh16 + j * 8);
    *(v8h*)(ko + j * 8) = *(const v8h*)(kh16 + j * 8);
  }
  _Float16* vo = V16 + h * DHEAD * NPIX;
#pragma unroll
  for (int d = 0; d < DHEAD; d++) vo[d * NPIX + n] = (_Float16)vp[d * NPIX];
}

// ---------------------------------------------------------------------------
// Stage 3: flash attention, one wave per 16-query tile, transposed scores
// with PERMUTED key-row assignment so the exp'd probabilities land directly
// in the B-operand layout of the PV WMMA (no cross-lane traffic in the loop).
//
//   St0 rows: keys {kb+0..7, kb+16..23}; St1 rows: keys {kb+8..15, kb+24..31}
//   => lane (q=lm, hi) holds, in St0/St1 regs 0..7, exactly the 16 K-slots
//      (keys kb+16*hi .. kb+16*hi+15, in order) that the B-tile layout
//      assigns to its half-wave.  b_p = pure in-lane f32->f16 pack.
//
//   p = exp2(st*t*log2e - |t|*log2e)  (exact: Q,K unit vectors => |logit|<=|t|)
//   O = V_tile(16d x 32k) x P^T(32k x 16q); per-query softmax denominator
//   closes with a single xor-16 after the loop.
// ---------------------------------------------------------------------------
__global__ __launch_bounds__(128) void attn_kernel(
    const _Float16* __restrict__ Q16,   // [2][4096][32]
    const _Float16* __restrict__ K16,   // [2][4096][32]
    const _Float16* __restrict__ V16,   // [2][32][4096]
    const float* __restrict__ tptr,     // [2] per-head temperature
    float* __restrict__ out)            // [2][32][4096]
{
  const int wave = threadIdx.x >> 5;
  const int lane = threadIdx.x & 31;
  const int lm = lane & 15;
  const int hi = lane >> 4;
  const int head = blockIdx.y;
  const int q0 = (blockIdx.x * 4 + wave) * 16;
  const float t = tptr[head];
  const float sc = t * LOG2E;            // exp2 scale
  const float sb = -fabsf(t) * LOG2E;    // constant shift (|logit| <= |t|)

  const _Float16* Qh = Q16 + head * NPIX * DHEAD;
  const _Float16* Kh = K16 + head * NPIX * DHEAD;
  const _Float16* Vh = V16 + head * DHEAD * NPIX;

  // B operand: Q tile, column q = q0+lm, d = hi*16 + i. One 32B load.
  const v16h b_q = *(const v16h*)(Qh + (q0 + lm) * DHEAD + hi * 16);

  // permuted key-row offset: lm<8 -> lm (keys j / j+8), lm>=8 -> lm+8
  const int krow = lm + (lm & 8);

  v8f o_lo = {}, o_hi = {};   // O rows d=0..15 / d=16..31
  float rsum = 0.f;

#pragma unroll 2
  for (int kb = 0; kb < NPIX; kb += 32) {
    // prefetch next key block into near caches
    __builtin_prefetch(Kh + (kb + 32 + lm) * DHEAD, 0, 3);
    __builtin_prefetch(Vh + lm * NPIX + kb + 32, 0, 3);

    // A operands (scores): St0 rows = keys kb+krow, St1 rows = keys kb+krow+8
    const _Float16* kr0 = Kh + (kb + krow) * DHEAD;
    const _Float16* kr1 = kr0 + 8 * DHEAD;
    v16h a_k0, a_k1;
    {
      v8h u0 = *(const v8h*)(kr0 + hi * 16);
      v8h u1 = *(const v8h*)(kr0 + hi * 16 + 8);
      v8h u2 = *(const v8h*)(kr1 + hi * 16);
      v8h u3 = *(const v8h*)(kr1 + hi * 16 + 8);
#pragma unroll
      for (int i = 0; i < 8; i++) {
        a_k0[i] = u0[i]; a_k0[8 + i] = u1[i];
        a_k1[i] = u2[i]; a_k1[8 + i] = u3[i];
      }
    }
    v8f z = {};
    v8f s0 = __builtin_amdgcn_wmma_f32_16x16x32_f16(false, a_k0, false, b_q,
                                                    (short)0, z, false, false);
    v8f s1 = __builtin_amdgcn_wmma_f32_16x16x32_f16(false, a_k1, false, b_q,
                                                    (short)0, z, false, false);

    // p = exp2(s*sc + sb); regs r of St0/St1 are this lane's B K-slots
    // (key kb+16*hi+r and kb+16*hi+8+r respectively), already in order.
    v16h b_p;
#pragma unroll
    for (int r = 0; r < 8; r++) {
      float p0 = __builtin_amdgcn_exp2f(s0[r] * sc + sb);
      float p1 = __builtin_amdgcn_exp2f(s1[r] * sc + sb);
      rsum += p0 + p1;
      b_p[r]     = (_Float16)p0;
      b_p[8 + r] = (_Float16)p1;
    }

    // A operands: V rows d=lm (o_lo) / 16+lm (o_hi), keys of this block.
    const _Float16* vr0 = Vh + lm * NPIX + kb;
    const _Float16* vr1 = Vh + (16 + lm) * NPIX + kb;
    v16h a_v0, a_v1;
    {
      v8h u0 = *(const v8h*)(vr0 + hi * 16);
      v8h u1 = *(const v8h*)(vr0 + hi * 16 + 8);
      v8h u2 = *(const v8h*)(vr1 + hi * 16);
      v8h u3 = *(const v8h*)(vr1 + hi * 16 + 8);
#pragma unroll
      for (int i = 0; i < 8; i++) {
        a_v0[i] = u0[i]; a_v0[8 + i] = u1[i];
        a_v1[i] = u2[i]; a_v1[8 + i] = u3[i];
      }
    }
    o_lo = __builtin_amdgcn_wmma_f32_16x16x32_f16(false, a_v0, false, b_p,
                                                  (short)0, o_lo, false, false);
    o_hi = __builtin_amdgcn_wmma_f32_16x16x32_f16(false, a_v1, false, b_p,
                                                  (short)0, o_hi, false, false);
  }

  // lanes q and q+16 hold keys 0..15 / 16..31 partials for query q=lm:
  // one xor-16 completes the softmax denominator.
  rsum += __shfl_xor(rsum, 16, 32);
  const float inv = 1.0f / rsum;

  const int q = q0 + lm;
  float* op = out + head * DHEAD * NPIX + q;
#pragma unroll
  for (int r = 0; r < 8; r++) {
    op[(r + 8 * hi) * NPIX]      = o_lo[r] * inv;
    op[(16 + r + 8 * hi) * NPIX] = o_hi[r] * inv;
  }
}

// ---------------------------------------------------------------------------
// Stage 4: 'B C X (H W) -> B (X C) H W' gather + depthwise 3x3 + LeakyReLU.
// ---------------------------------------------------------------------------
__global__ __launch_bounds__(256) void fus_kernel(
    const float* __restrict__ src,   // [2][32][4096]
    const float* __restrict__ w,     // [64][1][3][3]
    const float* __restrict__ bias,  // [64]
    float* __restrict__ out)         // [64][64][64]
{
  int idx = blockIdx.x * 256 + threadIdx.x;   // 64*4096
  int cc = idx >> 12, pix = idx & 4095;
  int h = cc & 1, dd = cc >> 1;
  const float* plane = src + h * DHEAD * NPIX + dd * NPIX;
  int y = pix >> 6, x = pix & 63;
  float acc = bias[cc];
#pragma unroll
  for (int ky = 0; ky < 3; ky++) {
    int iy = y + ky - 1;
    if (iy < 0 || iy > 63) continue;
#pragma unroll
    for (int kx = 0; kx < 3; kx++) {
      int ix = x + kx - 1;
      if (ix < 0 || ix > 63) continue;
      acc += w[cc * 9 + ky * 3 + kx] * plane[iy * 64 + ix];
    }
  }
  out[idx] = LRELU(acc);
}

// ---------------------------------------------------------------------------
extern "C" void kernel_launch(void* const* d_in, const int* in_sizes, int n_in,
                              void* d_out, int out_size, void* d_ws, size_t ws_size,
                              hipStream_t stream) {
  const float* x     = (const float*)d_in[0];
  const float* ir    = (const float*)d_in[1];
  const float* w_qkv = (const float*)d_in[2];
  const float* b_qkv = (const float*)d_in[3];
  const float* in_w  = (const float*)d_in[4];
  const float* in_b  = (const float*)d_in[5];
  const float* w_k   = (const float*)d_in[6];
  const float* b_k   = (const float*)d_in[7];
  const float* w_v   = (const float*)d_in[8];
  const float* b_v   = (const float*)d_in[9];
  const float* t     = (const float*)d_in[10];
  const float* t2    = (const float*)d_in[11];
  const float* w_fus = (const float*)d_in[12];
  const float* b_fus = (const float*)d_in[13];

  const size_t PLANE = (size_t)HEADS * DHEAD * NPIX;   // 262144 elements
  float* qv  = (float*)d_ws;
  float* kv  = qv  + PLANE;
  float* vv  = kv  + PLANE;
  float* kir = vv  + PLANE;
  float* vir = kir + PLANE;
  float* kf  = vir + PLANE;
  float* vf  = kf  + PLANE;
  float* at1 = vf  + PLANE;
  float* at2 = at1 + PLANE;
  _Float16* Q16 = (_Float16*)(at2 + PLANE);
  _Float16* K16 = Q16 + PLANE;
  _Float16* V16 = K16 + PLANE;
  // total ws: 9 f32 planes (9 MB) + 3 f16 planes (1.5 MB) = 10.5 MB

  // qkv for x (q,k,v) and ir (k,v only)
  qkv_kernel<<<192, 256, 0, stream>>>(x,  w_qkv, b_qkv, in_w, in_b, qv, kv, vv);
  qkv_kernel<<<192, 256, 0, stream>>>(ir, w_qkv, b_qkv, in_w, in_b,
                                      nullptr, kir, vir);
  // cross-modal K/V fusion convs
  fuse_kernel<<<1024, 256, 0, stream>>>(kv, kir, w_k, b_k, kf);
  fuse_kernel<<<1024, 256, 0, stream>>>(vv, vir, w_v, b_v, vf);

  // attention 1: attnfun(qv, kv, vv, t)
  prep_kernel<<<32, 256, 0, stream>>>(qv, kv, vv, Q16, K16, V16);
  attn_kernel<<<dim3(64, 2), 128, 0, stream>>>(Q16, K16, V16, t, at1);

  // attention 2: attnfun(at1, kf, vf, t2)
  prep_kernel<<<32, 256, 0, stream>>>(at1, kf, vf, Q16, K16, V16);
  attn_kernel<<<dim3(64, 2), 128, 0, stream>>>(Q16, K16, V16, t2, at2);

  // final transpose + depthwise fuse conv
  fus_kernel<<<1024, 256, 0, stream>>>(at2, w_fus, b_fus, (float*)d_out);
}
